// GraphSeqLM_63986422776181
// MI455X (gfx1250) — compile-verified, matrix operand order checked by reading, not run on hardware
//
#include <hip/hip_runtime.h>
#include <hip/hip_bf16.h>
#include <math.h>

// ---------------------------------------------------------------------------
// GraphSeqLM forward for MI455X (gfx1250, wave32, WMMA).
// Dense GEMMs -> v_wmma_f32_16x16x32_f16 (f32 accumulate), 2x2 tiles/wave,
// branch-free b128 operand loads from pre-transposed weights.
// Edge softmax/aggregation -> scatter kernels with atomics (L2-resident).
// ---------------------------------------------------------------------------

typedef __attribute__((ext_vector_type(16))) _Float16 v16h;
typedef __attribute__((ext_vector_type(8)))  float    v8f;

#define NODES_PER_GRAPH 2400
#define HEADS 4
#define CH 128            // HEADS * 32
#define BN_EPS 1e-5f
#define INV_SQRT32 0.17677669529663687f

// Load 16 fp32 -> v16h fragment. Lane layout for 16-bit A/B operands:
// elements 0..7  = p[0..7]   (K = half*8 + 0..7   within the 32-K step)
// elements 8..15 = p[16..23] (K = 16 + half*8 + 0..7)
__device__ __forceinline__ v16h load16h(const float* __restrict__ p) {
    float4 x0 = ((const float4*)p)[0];
    float4 x1 = ((const float4*)p)[1];
    float4 x2 = ((const float4*)(p + 16))[0];
    float4 x3 = ((const float4*)(p + 16))[1];
    v16h v;
    v[0]  = (_Float16)x0.x; v[1]  = (_Float16)x0.y;
    v[2]  = (_Float16)x0.z; v[3]  = (_Float16)x0.w;
    v[4]  = (_Float16)x1.x; v[5]  = (_Float16)x1.y;
    v[6]  = (_Float16)x1.z; v[7]  = (_Float16)x1.w;
    v[8]  = (_Float16)x2.x; v[9]  = (_Float16)x2.y;
    v[10] = (_Float16)x2.z; v[11] = (_Float16)x2.w;
    v[12] = (_Float16)x3.x; v[13] = (_Float16)x3.y;
    v[14] = (_Float16)x3.z; v[15] = (_Float16)x3.w;
    return v;
}

// ------------------------- WMMA GEMM ---------------------------------------
// C[M,Nc] = A[M,Kd] @ W[Kd,Nc] + bias, with WT = W^T stored [Nc,Kd].
// One wave computes a 32x32 C tile as 2x2 WMMA tiles (4 v_wmma / K-step).
// Requires: Kd % 32 == 0, M % 32 == 0, Nc % 32 == 0, 16B-aligned rows.
__global__ __launch_bounds__(32)
void wmma_gemm_bias(const float* __restrict__ A, const float* __restrict__ WT,
                    const float* __restrict__ bias, float* __restrict__ C,
                    int M, int Kd, int Nc)
{
    const int lane = threadIdx.x & 31;
    const int half = lane >> 4;
    const int l16  = lane & 15;
    const int row0 = blockIdx.x << 5;
    const int col0 = blockIdx.y << 5;

    const float* arow0 = A  + (size_t)(row0 + l16) * Kd + half * 8;
    const float* arow1 = arow0 + (size_t)16 * Kd;
    const float* brow0 = WT + (size_t)(col0 + l16) * Kd + half * 8;
    const float* brow1 = brow0 + (size_t)16 * Kd;

    v8f acc00 = {}, acc01 = {}, acc10 = {}, acc11 = {};
    for (int k0 = 0; k0 < Kd; k0 += 32) {
        v16h a0 = load16h(arow0 + k0);
        v16h a1 = load16h(arow1 + k0);
        v16h b0 = load16h(brow0 + k0);
        v16h b1 = load16h(brow1 + k0);
        acc00 = __builtin_amdgcn_wmma_f32_16x16x32_f16(false, a0, false, b0,
                                                       (short)0, acc00, false, false);
        acc01 = __builtin_amdgcn_wmma_f32_16x16x32_f16(false, a0, false, b1,
                                                       (short)0, acc01, false, false);
        acc10 = __builtin_amdgcn_wmma_f32_16x16x32_f16(false, a1, false, b0,
                                                       (short)0, acc10, false, false);
        acc11 = __builtin_amdgcn_wmma_f32_16x16x32_f16(false, a1, false, b1,
                                                       (short)0, acc11, false, false);
    }

    const float bb0 = bias[col0 + l16];
    const float bb1 = bias[col0 + 16 + l16];
#pragma unroll
    for (int r = 0; r < 8; ++r) {
        // C/D layout: lane col = l16 (+16 for second N tile), row = r + 8*half
        const size_t ra = (size_t)(row0 + r + 8 * half);
        const size_t rb = ra + 16;
        C[ra * Nc + col0 + l16]      = acc00[r] + bb0;
        C[ra * Nc + col0 + 16 + l16] = acc01[r] + bb1;
        C[rb * Nc + col0 + l16]      = acc10[r] + bb0;
        C[rb * Nc + col0 + 16 + l16] = acc11[r] + bb1;
    }
}

// ------------------------- weight transpose (+K pad) ------------------------
// WT[n*Kpad + k] = (k < Kd) ? W[k*Nc + n] : 0
__global__ void transpose_pad(const float* __restrict__ W, float* __restrict__ WT,
                              int Kd, int Nc, int Kpad) {
    size_t t = (size_t)blockIdx.x * blockDim.x + threadIdx.x;
    if (t >= (size_t)Nc * Kpad) return;
    int n = (int)(t / Kpad), k = (int)(t % Kpad);
    WT[t] = (k < Kd) ? W[(size_t)k * Nc + n] : 0.f;
}

// ------------------------- helpers -----------------------------------------
__device__ __forceinline__ unsigned f2ord(float f) {
    unsigned u = __float_as_uint(f);
    return (u & 0x80000000u) ? ~u : (u | 0x80000000u);
}
__device__ __forceinline__ float ord2f(unsigned u) {
    unsigned v = (u & 0x80000000u) ? (u & 0x7FFFFFFFu) : ~u;
    return __uint_as_float(v);
}

__global__ void fill_u32(unsigned* p, unsigned v, size_t n) {
    size_t t = (size_t)blockIdx.x * blockDim.x + threadIdx.x;
    if (t < n) p[t] = v;
}
__global__ void fill_f32(float* p, float v, size_t n) {
    size_t t = (size_t)blockIdx.x * blockDim.x + threadIdx.x;
    if (t < n) p[t] = v;
}

// cat = [x (N,8) | seq tiled over batch (N,128) | zero pad] -> (N,160)
__global__ void build_cat(const float* __restrict__ x, const float* __restrict__ seq,
                          float* __restrict__ cat, int N) {
    size_t t = (size_t)blockIdx.x * blockDim.x + threadIdx.x;
    if (t >= (size_t)N * 160) return;
    int i = (int)(t / 160), j = (int)(t % 160);
    float v;
    if (j < 8)        v = x[(size_t)i * 8 + j];
    else if (j < 136) v = seq[(size_t)(i % NODES_PER_GRAPH) * CH + (j - 8)];
    else              v = 0.f;
    cat[t] = v;
}

// cat2 = [xglm (N,32) | xi (N,32)] -> (N,64)
__global__ void build_cat2(const float* __restrict__ xglm, const float* __restrict__ xi,
                           float* __restrict__ cat2, int N) {
    size_t t = (size_t)blockIdx.x * blockDim.x + threadIdx.x;
    if (t >= (size_t)N * 64) return;
    int i = (int)(t / 64), j = (int)(t % 64);
    cat2[t] = (j < 32) ? xglm[(size_t)i * 32 + j] : xi[(size_t)i * 32 + (j - 32)];
}

// ------------------------- edge passes -------------------------------------
// thread per (edge, head)
__global__ void edge_alpha(const int* __restrict__ src, const int* __restrict__ dst, int E,
                           const float* __restrict__ Q, const float* __restrict__ Kf,
                           float* __restrict__ alpha, unsigned* __restrict__ amax) {
    size_t t = (size_t)blockIdx.x * blockDim.x + threadIdx.x;
    if (t >= (size_t)E * HEADS) return;
    int e = (int)(t >> 2), h = (int)(t & 3);
    int s = src[e], d = dst[e];
    const float4* q4 = (const float4*)(Q  + (size_t)d * CH + h * 32);
    const float4* k4 = (const float4*)(Kf + (size_t)s * CH + h * 32);
    float acc = 0.f;
#pragma unroll
    for (int i = 0; i < 8; ++i) {
        float4 a = q4[i], b = k4[i];
        acc += a.x * b.x + a.y * b.y + a.z * b.z + a.w * b.w;
    }
    acc *= INV_SQRT32;
    alpha[t] = acc;
    atomicMax(&amax[(size_t)d * HEADS + h], f2ord(acc));
}

__global__ void edge_exp(const int* __restrict__ dst, int E,
                         float* __restrict__ alpha, const unsigned* __restrict__ amax,
                         float* __restrict__ den) {
    size_t t = (size_t)blockIdx.x * blockDim.x + threadIdx.x;
    if (t >= (size_t)E * HEADS) return;
    int e = (int)(t >> 2), h = (int)(t & 3);
    int d = dst[e];
    float m  = ord2f(amax[(size_t)d * HEADS + h]);
    float ev = expf(alpha[t] - m);
    alpha[t] = ev;
    atomicAdd(&den[(size_t)d * HEADS + h], ev);
}

// out already holds the root/skip term (x @ Ws + bs); scatter-add messages.
__global__ void edge_scatter(const int* __restrict__ src, const int* __restrict__ dst, int E,
                             const float* __restrict__ V, const float* __restrict__ alpha,
                             const float* __restrict__ den, float* __restrict__ out) {
    size_t t = (size_t)blockIdx.x * blockDim.x + threadIdx.x;
    if (t >= (size_t)E * HEADS) return;
    int e = (int)(t >> 2), h = (int)(t & 3);
    int s = src[e], d = dst[e];
    float a = alpha[t] / (den[(size_t)d * HEADS + h] + 1e-16f);
    const float4* v4 = (const float4*)(V + (size_t)s * CH + h * 32);
    float*        o  = out + (size_t)d * CH + h * 32;
#pragma unroll
    for (int i = 0; i < 8; ++i) {
        float4 vv = v4[i];
        atomicAdd(&o[4 * i + 0], vv.x * a);
        atomicAdd(&o[4 * i + 1], vv.y * a);
        atomicAdd(&o[4 * i + 2], vv.z * a);
        atomicAdd(&o[4 * i + 3], vv.w * a);
    }
}

// ------------------------- BatchNorm + LeakyReLU ---------------------------
__global__ void bn_stats(const float* __restrict__ X, int Nrows, int Cc,
                         float* __restrict__ mean, float* __restrict__ rstd) {
    __shared__ float s1[256], s2[256];
    int c = blockIdx.x;
    float sum = 0.f, sq = 0.f;
    for (int i = threadIdx.x; i < Nrows; i += blockDim.x) {
        float v = X[(size_t)i * Cc + c];
        sum += v; sq += v * v;
    }
    s1[threadIdx.x] = sum; s2[threadIdx.x] = sq;
    __syncthreads();
    for (int o = blockDim.x >> 1; o > 0; o >>= 1) {
        if (threadIdx.x < (unsigned)o) {
            s1[threadIdx.x] += s1[threadIdx.x + o];
            s2[threadIdx.x] += s2[threadIdx.x + o];
        }
        __syncthreads();
    }
    if (threadIdx.x == 0) {
        float m = s1[0] / Nrows;
        float var = s2[0] / Nrows - m * m;
        mean[c] = m;
        rstd[c] = rsqrtf(var + BN_EPS);
    }
}

__global__ void bn_apply_leaky(float* __restrict__ X, const float* __restrict__ g,
                               const float* __restrict__ b, const float* __restrict__ mean,
                               const float* __restrict__ rstd, int Nrows, int Cc) {
    size_t t = (size_t)blockIdx.x * blockDim.x + threadIdx.x;
    if (t >= (size_t)Nrows * Cc) return;
    int c = (int)(t % Cc);
    float y = g[c] * (X[t] - mean[c]) * rstd[c] + b[c];
    X[t] = (y > 0.f) ? y : 0.1f * y;
}

// ------------------------- pooling / classifier ----------------------------
__global__ void pool_mean(const float* __restrict__ H, float* __restrict__ pool,
                          int nodes, int Cc, int B) {
    int t = blockIdx.x * blockDim.x + threadIdx.x;
    if (t >= B * Cc) return;
    int b = t / Cc, c = t % Cc;
    const float* p = H + (size_t)b * nodes * Cc + c;
    float s = 0.f;
    for (int i = 0; i < nodes; ++i) s += p[(size_t)i * Cc];
    pool[t] = s / nodes;
}

__global__ void pred_k(const float* __restrict__ pool, const float* __restrict__ Wp,
                       const float* __restrict__ bp, float* __restrict__ out,
                       int B, int Cc, int NC) {
    int t = blockIdx.x * blockDim.x + threadIdx.x;
    if (t >= B * NC) return;
    int b = t / NC, k = t % NC;
    float s = bp[k];
    for (int c = 0; c < Cc; ++c) s += pool[(size_t)b * Cc + c] * Wp[(size_t)c * NC + k];
    out[t] = s;
}

// ------------------------- param flatten order -----------------------------
// Assumes params dict leaves arrive after the 6 positional inputs in
// jax.tree_util order (dict keys sorted, ASCII: 'W' < 'b' < 'c' < 'g' < 'i';
// conv sub-dicts flatten as Wk,Wq,Ws,Wv,bk,bq,bs,bv).
enum {
    P_Wdna = 0, P_Wglm, P_Wint, P_Wmod, P_Wpred, P_Wprot, P_Wrna,
    P_b_1, P_b_2, P_b_3, P_b_i1, P_b_i2, P_b_i3,
    P_bdna, P_bglm, P_bint, P_bmod, P_bpred, P_bprot, P_brna,
    P_c1_Wk, P_c1_Wq, P_c1_Ws, P_c1_Wv, P_c1_bk, P_c1_bq, P_c1_bs, P_c1_bv,
    P_c2_Wk, P_c2_Wq, P_c2_Ws, P_c2_Wv, P_c2_bk, P_c2_bq, P_c2_bs, P_c2_bv,
    P_c3_Wk, P_c3_Wq, P_c3_Ws, P_c3_Wv, P_c3_bk, P_c3_bq, P_c3_bs, P_c3_bv,
    P_g_1, P_g_2, P_g_3, P_g_i1, P_g_i2, P_g_i3,
    P_ic1_Wk, P_ic1_Wq, P_ic1_Ws, P_ic1_Wv, P_ic1_bk, P_ic1_bq, P_ic1_bs, P_ic1_bv,
    P_ic2_Wk, P_ic2_Wq, P_ic2_Ws, P_ic2_Wv, P_ic2_bk, P_ic2_bq, P_ic2_bs, P_ic2_bv,
    P_ic3_Wk, P_ic3_Wq, P_ic3_Ws, P_ic3_Wv, P_ic3_bk, P_ic3_bq, P_ic3_bs, P_ic3_bv
};

struct ConvT { float *q, *k, *v, *s; };

extern "C" void kernel_launch(void* const* d_in, const int* in_sizes, int n_in,
                              void* d_out, int out_size, void* d_ws, size_t ws_size,
                              hipStream_t stream) {
    const float* x    = (const float*)d_in[0];
    const int*   iedg = (const int*)d_in[1];
    const int*   aedg = (const int*)d_in[2];
    const float* dna  = (const float*)d_in[3];
    const float* rna  = (const float*)d_in[4];
    const float* prot = (const float*)d_in[5];

    const int N    = in_sizes[0] / 8;                 // 38400
    const int B    = N / NODES_PER_GRAPH;             // 16
    const int Eint = in_sizes[1] / 2;                 // 320000
    const int Eall = in_sizes[2] / 2;                 // 640000
    const int Emax = (Eint > Eall) ? Eint : Eall;

#define PP(i) ((const float*)d_in[6 + (i)])

    // -------- workspace layout (floats) --------
    float* w = (float*)d_ws;
    size_t off = 0;
    auto alloc = [&](size_t n) { float* p = w + off; off += n; return p; };
    float* SEQ   = alloc((size_t)NODES_PER_GRAPH * CH);
    float* XGLM  = alloc((size_t)N * 32);
    float* CAT   = alloc((size_t)N * 160);    // later reused for CAT2/XI/XC
    float* H0    = alloc((size_t)N * CH);
    float* H1    = alloc((size_t)N * CH);
    float* Qb    = alloc((size_t)N * CH);
    float* Kbuf  = alloc((size_t)N * CH);
    float* Vb    = alloc((size_t)N * CH);
    float* ALPHA = alloc((size_t)Emax * HEADS);
    unsigned* AMAX = (unsigned*)alloc((size_t)N * HEADS);
    float* DEN   = alloc((size_t)N * HEADS);
    float* MEAN  = alloc(CH);
    float* RSTD  = alloc(CH);
    float* POOL  = alloc((size_t)B * CH);
    float* CAT2  = CAT;                        // (N,64)
    float* XI    = CAT + (size_t)N * 64;       // (N,32)
    float* XC    = CAT + (size_t)N * 96;       // (N,32)

    auto xpose = [&](const float* W_, float* WT_, int Kd, int Nc, int Kpad) {
        size_t tot = (size_t)Nc * Kpad;
        transpose_pad<<<(unsigned)((tot + 255) / 256), 256, 0, stream>>>(
            W_, WT_, Kd, Nc, Kpad);
    };
    auto tw = [&](int pidx, int Kd, int Nc, int Kpad) {
        float* p = alloc((size_t)Nc * Kpad);
        xpose(PP(pidx), p, Kd, Nc, Kpad);
        return p;
    };

    // -------- pre-transpose (and K-pad) all WMMA weights --------
    float* T_Wdna  = tw(P_Wdna,  768, CH, 768);
    float* T_Wrna  = tw(P_Wrna,  768, CH, 768);
    float* T_Wprot = tw(P_Wprot, 768, CH, 768);
    float* T_Wglm  = tw(P_Wglm,  136, 32, 160);
    float* T_Wint  = tw(P_Wint,  CH,  32, CH);
    float* T_Wmod  = tw(P_Wmod,  64,  32, 64);
    auto twconv = [&](int Kd, int pWq, int pWk, int pWv, int pWs) {
        ConvT c;
        c.q = tw(pWq, Kd, CH, Kd);
        c.k = tw(pWk, Kd, CH, Kd);
        c.v = tw(pWv, Kd, CH, Kd);
        c.s = tw(pWs, Kd, CH, Kd);
        return c;
    };
    ConvT Tic1 = twconv(32, P_ic1_Wq, P_ic1_Wk, P_ic1_Wv, P_ic1_Ws);
    ConvT Tic2 = twconv(CH, P_ic2_Wq, P_ic2_Wk, P_ic2_Wv, P_ic2_Ws);
    ConvT Tic3 = twconv(CH, P_ic3_Wq, P_ic3_Wk, P_ic3_Wv, P_ic3_Ws);
    ConvT Tc1  = twconv(32, P_c1_Wq,  P_c1_Wk,  P_c1_Wv,  P_c1_Ws);
    ConvT Tc2  = twconv(CH, P_c2_Wq,  P_c2_Wk,  P_c2_Wv,  P_c2_Ws);
    ConvT Tc3  = twconv(CH, P_c3_Wq,  P_c3_Wk,  P_c3_Wv,  P_c3_Ws);

    auto gemm = [&](const float* A, const float* WT_, const float* b_, float* C_,
                    int M, int Kd, int Nc) {
        dim3 g(M / 32, Nc / 32);
        wmma_gemm_bias<<<g, dim3(32), 0, stream>>>(A, WT_, b_, C_, M, Kd, Nc);
    };

    auto tconv = [&](const float* in, int Kin, const int* ei, int E, const ConvT& W_,
                     const float* bq, const float* bk, const float* bv, const float* bs,
                     float* out) {
        const int* src = ei;
        const int* dst = ei + E;
        gemm(in, W_.q, bq, Qb,   N, Kin, CH);
        gemm(in, W_.k, bk, Kbuf, N, Kin, CH);
        gemm(in, W_.v, bv, Vb,   N, Kin, CH);
        gemm(in, W_.s, bs, out,  N, Kin, CH);   // root/skip; messages add on top
        size_t nh = (size_t)N * HEADS;
        fill_u32<<<(unsigned)((nh + 255) / 256), 256, 0, stream>>>(AMAX, 0u, nh);
        fill_f32<<<(unsigned)((nh + 255) / 256), 256, 0, stream>>>(DEN, 0.f, nh);
        size_t eh = (size_t)E * HEADS;
        unsigned gb = (unsigned)((eh + 255) / 256);
        edge_alpha<<<gb, 256, 0, stream>>>(src, dst, E, Qb, Kbuf, ALPHA, AMAX);
        edge_exp<<<gb, 256, 0, stream>>>(dst, E, ALPHA, AMAX, DEN);
        edge_scatter<<<gb, 256, 0, stream>>>(src, dst, E, Vb, ALPHA, DEN, out);
    };

    auto bnlrelu = [&](float* X, const float* g_, const float* b_) {
        bn_stats<<<CH, 256, 0, stream>>>(X, N, CH, MEAN, RSTD);
        size_t tot = (size_t)N * CH;
        bn_apply_leaky<<<(unsigned)((tot + 255) / 256), 256, 0, stream>>>(
            X, g_, b_, MEAN, RSTD, N, CH);
    };

    // -------- 1. LM projections -> SEQ (2400,128) --------
    gemm(dna,  T_Wdna,  PP(P_bdna),  SEQ,             800, 768, CH);
    gemm(rna,  T_Wrna,  PP(P_brna),  SEQ +  800 * CH, 800, 768, CH);
    gemm(prot, T_Wprot, PP(P_bprot), SEQ + 1600 * CH, 800, 768, CH);

    // -------- 2. x_glm = [x | seq] @ Wglm + bglm --------
    {
        size_t tot = (size_t)N * 160;
        build_cat<<<(unsigned)((tot + 255) / 256), 256, 0, stream>>>(x, SEQ, CAT, N);
    }
    gemm(CAT, T_Wglm, PP(P_bglm), XGLM, N, 160, 32);

    // -------- 3. internal conv stack --------
    tconv(XGLM, 32, iedg, Eint, Tic1,
          PP(P_ic1_bq), PP(P_ic1_bk), PP(P_ic1_bv), PP(P_ic1_bs), H0);
    bnlrelu(H0, PP(P_g_i1), PP(P_b_i1));
    tconv(H0, CH, iedg, Eint, Tic2,
          PP(P_ic2_bq), PP(P_ic2_bk), PP(P_ic2_bv), PP(P_ic2_bs), H1);
    bnlrelu(H1, PP(P_g_i2), PP(P_b_i2));
    tconv(H1, CH, iedg, Eint, Tic3,
          PP(P_ic3_bq), PP(P_ic3_bk), PP(P_ic3_bv), PP(P_ic3_bs), H0);
    bnlrelu(H0, PP(P_g_i3), PP(P_b_i3));

    // -------- 4. modality fusion --------
    gemm(H0, T_Wint, PP(P_bint), XI, N, CH, 32);
    {
        size_t tot = (size_t)N * 64;
        build_cat2<<<(unsigned)((tot + 255) / 256), 256, 0, stream>>>(XGLM, XI, CAT2, N);
    }
    gemm(CAT2, T_Wmod, PP(P_bmod), XC, N, 64, 32);

    // -------- 5. all-edge conv stack --------
    tconv(XC, 32, aedg, Eall, Tc1,
          PP(P_c1_bq), PP(P_c1_bk), PP(P_c1_bv), PP(P_c1_bs), H1);
    bnlrelu(H1, PP(P_g_1), PP(P_b_1));
    tconv(H1, CH, aedg, Eall, Tc2,
          PP(P_c2_bq), PP(P_c2_bk), PP(P_c2_bv), PP(P_c2_bs), H0);
    bnlrelu(H0, PP(P_g_2), PP(P_b_2));
    tconv(H0, CH, aedg, Eall, Tc3,
          PP(P_c3_bq), PP(P_c3_bk), PP(P_c3_bv), PP(P_c3_bs), H1);
    bnlrelu(H1, PP(P_g_3), PP(P_b_3));

    // -------- 6. pool + classifier --------
    pool_mean<<<(B * CH + 255) / 256, 256, 0, stream>>>(H1, POOL, NODES_PER_GRAPH, CH, B);
    pred_k<<<1, 64, 0, stream>>>(POOL, PP(P_Wpred), PP(P_bpred), (float*)d_out, B, CH, 2);

#undef PP
    (void)n_in; (void)out_size; (void)ws_size;
}
